// MHA_17635135717660
// MI455X (gfx1250) — matrix operand order repaired and checked by reading, not compiled, not run on hardware
//
#include <hip/hip_runtime.h>
#include <stdint.h>

#ifndef __has_builtin
#define __has_builtin(x) 0
#endif

typedef __bf16 bf16;
typedef __attribute__((ext_vector_type(16))) __bf16        v16bf;
typedef __attribute__((ext_vector_type(8)))  float         v8f;
typedef __attribute__((ext_vector_type(4)))  unsigned int  u32x4;
typedef __attribute__((ext_vector_type(8)))  int           i32x8;
typedef __attribute__((ext_vector_type(4)))  int           i32x4;

#define D_MODEL 512
#define NH      8
#define DK      64
#define SEQ     2048
#define BATCH   4
#define MTOT    (BATCH*SEQ)   // 8192 rows
#define LN_EPS  1e-5f
#define KT      64            // GEMM K-slab per LDS stage

// ---------------------------------------------------------------- WMMA core
static __device__ __forceinline__ v8f wmma_bf16f32(v16bf a, v16bf b, v8f c) {
  // v_wmma_f32_16x16x32_bf16  D = A(16x32) * B(32x16) + C
  return __builtin_amdgcn_wmma_f32_16x16x32_bf16(false, a, false, b, (short)0, c,
                                                 false, false);
}
static __device__ __forceinline__ v8f vzero8() {
  v8f z = {0.f,0.f,0.f,0.f,0.f,0.f,0.f,0.f};
  return z;
}

// A fragment: 16x32 bf16 tile at (row0, k0) of a row-major matrix, leading dim ld.
// ISA layout: lane = 16*half + M; lanes 0-15 hold K {0..7,16..23}, lanes 16-31 hold
// K {8..15,24..31}; pairs per VGPR.
static __device__ __forceinline__ v16bf a_frag(const bf16* src, int ld, int row0,
                                               int k0, int lane) {
  const bf16* p = src + (size_t)(row0 + (lane & 15)) * ld + k0 + ((lane & 16) ? 8 : 0);
  v16bf a;
#pragma unroll
  for (int v = 0; v < 8; ++v) {
    int k = (v < 4) ? 2 * v : 2 * v + 8;
    a[2 * v]     = p[k];
    a[2 * v + 1] = p[k + 1];
  }
  return a;
}

// B fragment: 32x16 (KxN). Source stored [col][k] (K contiguous, leading dim ld).
// lane = 16*half + N; lanes 0-15 hold K {0..7,16..23}, lanes 16-31 hold K {8..15,24..31}.
static __device__ __forceinline__ v16bf b_frag(const bf16* src, int ld, int col0,
                                               int k0, int lane) {
  const bf16* p = src + (size_t)(col0 + (lane & 15)) * ld + k0 + ((lane & 16) ? 8 : 0);
  v16bf b;
#pragma unroll
  for (int i = 0; i < 16; ++i) b[i] = p[(i < 8) ? i : i + 8];
  return b;
}

// ---------------------------------------------------------------- TDM staging
#if __has_builtin(__builtin_amdgcn_tensor_load_to_lds)
#define HAVE_TDM 1
// DMA a [tile_d1 rows x tile_d0 elems] bf16 tile (row stride stride0 elems) into LDS.
static __device__ __forceinline__ void tdm_load_2d_bf16(unsigned lds_addr,
    const bf16* gptr, unsigned tensor_d0, unsigned tensor_d1,
    unsigned tile_d0, unsigned tile_d1, unsigned stride0) {
  unsigned long long ga = (unsigned long long)(uintptr_t)gptr;
  u32x4 g0;
  g0[0] = 1u;                                                  // count=1, user D#
  g0[1] = lds_addr;                                            // LDS byte address
  g0[2] = (unsigned)ga;                                        // global_addr[31:0]
  g0[3] = (unsigned)((ga >> 32) & 0x01FFFFFFu) | (2u << 30);   // addr[56:32] | type=2
  i32x8 g1;
  g1[0] = (int)(1u << 16);                                     // data_size=1 (2B)
  g1[1] = (int)((tensor_d0 & 0xFFFFu) << 16);                  // tensor_dim0[15:0]
  g1[2] = (int)((tensor_d0 >> 16) | ((tensor_d1 & 0xFFFFu) << 16));
  g1[3] = (int)((tensor_d1 >> 16) | (tile_d0 << 16));          // .. | tile_dim0
  g1[4] = (int)(tile_d1 & 0xFFFFu);                            // tile_dim1, tile_dim2=0
  g1[5] = (int)stride0;                                        // dim0_stride[31:0]
  g1[6] = 0;
  g1[7] = 0;
  i32x4 z4 = {0, 0, 0, 0};
#if defined(__clang_major__) && (__clang_major__ >= 23)
  i32x8 z8 = {0, 0, 0, 0, 0, 0, 0, 0};
  __builtin_amdgcn_tensor_load_to_lds(g0, g1, z4, z4, z8, 0);
#else
  __builtin_amdgcn_tensor_load_to_lds(g0, g1, z4, z4, 0);
#endif
}
#else
#define HAVE_TDM 0
#endif

// ---------------------------------------------------------------- kernels
// Fold rank-4 LoRA update into the frozen weights, convert all weights to bf16.
// W_eff[o,i] = W[o,i] + sum_r B[o,r]*A[r,i]   (alpha/rank == 1.0)
__global__ void lora_prep_weights(const float* __restrict__ Wq, const float* __restrict__ Wk,
                                  const float* __restrict__ Wv, const float* __restrict__ Aq,
                                  const float* __restrict__ Bq, const float* __restrict__ Ak,
                                  const float* __restrict__ Bk, const float* __restrict__ Av,
                                  const float* __restrict__ Bv, const float* __restrict__ Wo,
                                  bf16* __restrict__ WqE, bf16* __restrict__ WkE,
                                  bf16* __restrict__ WvE, bf16* __restrict__ WoB) {
  int idx = blockIdx.x * blockDim.x + threadIdx.x;
  if (idx >= D_MODEL * D_MODEL) return;
  int o = idx >> 9, i = idx & 511;
  float lq = 0.f, lk = 0.f, lv = 0.f;
#pragma unroll
  for (int r = 0; r < 4; ++r) {
    lq += Bq[o * 4 + r] * Aq[r * D_MODEL + i];
    lk += Bk[o * 4 + r] * Ak[r * D_MODEL + i];
    lv += Bv[o * 4 + r] * Av[r * D_MODEL + i];
  }
  WqE[idx] = (bf16)(Wq[idx] + lq);
  WkE[idx] = (bf16)(Wk[idx] + lk);
  WvE[idx] = (bf16)(Wv[idx] + lv);
  WoB[idx] = (bf16)Wo[idx];
}

// C[M,512] = X[M,512] @ W^T, W stored [out=512][in=512] bf16 (K contiguous per row).
// WG tile 64(M) x 128(N), 8 waves, each wave 32x32 via 2x2 wmma accumulators.
// K-slab 64, double-buffered LDS; the weight tile for slab t+1 is DMA'd by the
// Tensor Data Mover while slab t's 8 WMMAs execute (s_wait_tensorcnt after compute).
// writeMode: 0 = bf16 [M,512]; 1 = fp32 [M,512]; 2 = bf16 transposed per batch
// ([b, col, s] with s-stride 1) for the attention PV operand.
__global__ void lora_gemm_xwT(const float* __restrict__ X, const bf16* __restrict__ W,
                              bf16* __restrict__ outBf, float* __restrict__ outF,
                              int writeMode, float outScale) {
  __shared__ bf16 ldsA[2][64 * KT];    // 2 x 8 KB
  __shared__ bf16 ldsB[2][128 * KT];   // 2 x 16 KB
  const int tid = threadIdx.x, lane = tid & 31, wave = tid >> 5;
  const int m0 = blockIdx.x * 64;
  const int n0 = blockIdx.y * 128;
  const int mBase = (wave & 1) * 32, nBase = (wave >> 1) * 32;
  v8f acc00 = vzero8(), acc01 = vzero8(), acc10 = vzero8(), acc11 = vzero8();

  const int ar = tid >> 2, ac = (tid & 3) * 16;   // A stage: 16 fp32 / thread

  auto stageA = [&](int buf, int k0) {
    const float* s = X + (size_t)(m0 + ar) * D_MODEL + k0 + ac;
    __builtin_prefetch(s + KT, 0, 3);             // next slab, near-cache hint
#pragma unroll
    for (int j = 0; j < 16; ++j) ldsA[buf][ar * KT + ac + j] = (bf16)s[j];
  };
#if HAVE_TDM
  auto stageB = [&](int buf, int k0) {
    if (wave == 0)
      tdm_load_2d_bf16((unsigned)(uintptr_t)(&ldsB[buf][0]),
                       W + (size_t)n0 * D_MODEL + k0,
                       (unsigned)D_MODEL, (unsigned)D_MODEL,
                       (unsigned)KT, 128u, (unsigned)D_MODEL);
  };
#else
  const int br = tid >> 1, bc = (tid & 1) * 32;   // B stage: 32 bf16 / thread
  auto stageB = [&](int buf, int k0) {
    const bf16* s = W + (size_t)(n0 + br) * D_MODEL + k0 + bc;
#pragma unroll
    for (int j = 0; j < 32; ++j) ldsB[buf][br * KT + bc + j] = s[j];
  };
#endif

  stageA(0, 0);
  stageB(0, 0);
#if HAVE_TDM
  if (wave == 0) __builtin_amdgcn_s_wait_tensorcnt(0);
#endif
  __syncthreads();

  const int NSLAB = D_MODEL / KT;                 // 8
  for (int kt = 0; kt < NSLAB; ++kt) {
    const int cur = kt & 1, nxt = cur ^ 1;
    if (kt + 1 < NSLAB) {                         // stage next slab (other buffer)
      stageA(nxt, (kt + 1) * KT);
      stageB(nxt, (kt + 1) * KT);                 // TDM overlaps the WMMAs below
    }
#pragma unroll
    for (int kk = 0; kk < KT; kk += 32) {
      v16bf a0 = a_frag(&ldsA[cur][0], KT, mBase,      kk, lane);
      v16bf a1 = a_frag(&ldsA[cur][0], KT, mBase + 16, kk, lane);
      v16bf b0 = b_frag(&ldsB[cur][0], KT, nBase,      kk, lane);
      v16bf b1 = b_frag(&ldsB[cur][0], KT, nBase + 16, kk, lane);
      acc00 = wmma_bf16f32(a0, b0, acc00);
      acc01 = wmma_bf16f32(a0, b1, acc01);
      acc10 = wmma_bf16f32(a1, b0, acc10);
      acc11 = wmma_bf16f32(a1, b1, acc11);
    }
#if HAVE_TDM
    if (wave == 0 && kt + 1 < NSLAB) __builtin_amdgcn_s_wait_tensorcnt(0);
#endif
    __syncthreads();                              // next slab staged & cur consumed
  }

  const int n = lane & 15, mh = (lane >> 4) << 3; // C/D layout: N=lane&15, M=r+8*half
#pragma unroll
  for (int r = 0; r < 8; ++r) {
    int r0 = m0 + mBase + mh + r, r1 = r0 + 16;
    int c0 = n0 + nBase + n,      c1 = c0 + 16;
    float v00 = acc00[r] * outScale, v01 = acc01[r] * outScale;
    float v10 = acc10[r] * outScale, v11 = acc11[r] * outScale;
    if (writeMode == 1) {
      outF[(size_t)r0 * D_MODEL + c0] = v00;
      outF[(size_t)r0 * D_MODEL + c1] = v01;
      outF[(size_t)r1 * D_MODEL + c0] = v10;
      outF[(size_t)r1 * D_MODEL + c1] = v11;
    } else if (writeMode == 0) {
      outBf[(size_t)r0 * D_MODEL + c0] = (bf16)v00;
      outBf[(size_t)r0 * D_MODEL + c1] = (bf16)v01;
      outBf[(size_t)r1 * D_MODEL + c0] = (bf16)v10;
      outBf[(size_t)r1 * D_MODEL + c1] = (bf16)v11;
    } else {  // transposed per batch: out[(b*512 + col)*SEQ + s]
      int b0r = r0 >> 11, s0 = r0 & (SEQ - 1);
      int b1r = r1 >> 11, s1 = r1 & (SEQ - 1);
      outBf[((size_t)(b0r * D_MODEL + c0)) * SEQ + s0] = (bf16)v00;
      outBf[((size_t)(b0r * D_MODEL + c1)) * SEQ + s0] = (bf16)v01;
      outBf[((size_t)(b1r * D_MODEL + c0)) * SEQ + s1] = (bf16)v10;
      outBf[((size_t)(b1r * D_MODEL + c1)) * SEQ + s1] = (bf16)v11;
    }
  }
}

// One workgroup per (b, h, 16-row q stripe). Scores stripe (16x2048 fp32 = 128 KB)
// stays resident in LDS (CDNA5: 320 KB/WGP) -> single-pass mask+softmax+PV; attn
// probabilities stream to d_out exactly once via non-temporal stores (537 MB,
// > 2x the 192 MB L2 -> do not cache).
__global__ void lora_attn_kernel(const bf16* __restrict__ Qbf, const bf16* __restrict__ Kbf,
                                 const bf16* __restrict__ VbfT, float* __restrict__ attnOut,
                                 float* __restrict__ ctxOut) {
  extern __shared__ float smem[];
  float* scores  = smem;               // 16 * 2048 fp32
  float* partial = smem + 16 * SEQ;    // 8 waves * 16 * 64 fp32

  const int tid = threadIdx.x, lane = tid & 31, wave = tid >> 5;
  const int bid = blockIdx.x;
  const int qt = bid & 127;            // S/16 stripes
  const int h  = (bid >> 7) & 7;
  const int b  = bid >> 10;
  const int q0 = qt * 16;
  const int L  = (qt + 1) * 16;        // causal valid length for this stripe
  const int Lpad = (L + 31) & ~31;
  const int n = lane & 15, mh = (lane >> 4) << 3;

  const bf16* Qb  = Qbf  + (size_t)b * SEQ * D_MODEL + h * DK;
  const bf16* Kb  = Kbf  + (size_t)b * SEQ * D_MODEL + h * DK;
  const bf16* VbT = VbfT + (size_t)(b * D_MODEL + h * DK) * SEQ;   // [d][s], s contig

  // Q stripe fragments (d = 0..31 and 32..63); 1/sqrt(d_k) already folded into Qbf.
  v16bf aq0 = a_frag(Qb, D_MODEL, q0, 0, lane);
  v16bf aq1 = a_frag(Qb, D_MODEL, q0, 32, lane);

  // ---- scores = Q K^T (only tiles on/below the diagonal)
  for (int kt = wave; kt <= qt; kt += 8) {
    int k0 = kt * 16;
    v16bf kb0 = b_frag(Kb, D_MODEL, k0, 0, lane);   // B[k=d][n=key-row]
    v16bf kb1 = b_frag(Kb, D_MODEL, k0, 32, lane);
    v8f acc = vzero8();
    acc = wmma_bf16f32(aq0, kb0, acc);
    acc = wmma_bf16f32(aq1, kb1, acc);
#pragma unroll
    for (int r = 0; r < 8; ++r) {
      int m = mh + r, kc = k0 + n;
      scores[m * SEQ + kc] = (kc > q0 + m) ? -1e9f : acc[r];
    }
  }
  __syncthreads();

  // ---- softmax over the valid prefix; stream probs to d_out; zero the masked tail
  float* attnBase = attnOut + ((size_t)((b * NH + h) * SEQ + q0)) * SEQ;
#pragma unroll
  for (int rr = 0; rr < 2; ++rr) {
    int m = wave * 2 + rr;
    float* srow = scores + m * SEQ;
    float mx = -3.0e38f;
    for (int k = lane; k < L; k += 32) mx = fmaxf(mx, srow[k]);
#pragma unroll
    for (int o = 16; o; o >>= 1) mx = fmaxf(mx, __shfl_xor(mx, o, 32));
    float sum = 0.f;
    for (int k = lane; k < L; k += 32) { float p = __expf(srow[k] - mx); srow[k] = p; sum += p; }
#pragma unroll
    for (int o = 16; o; o >>= 1) sum += __shfl_xor(sum, o, 32);
    float inv = 1.f / sum;
    float* arow = attnBase + (size_t)m * SEQ;
    for (int k = lane; k < L; k += 32) {
      float p = srow[k] * inv;
      srow[k] = p;
      __builtin_nontemporal_store(p, &arow[k]);
    }
    for (int k = L + lane; k < Lpad; k += 32) srow[k] = 0.f;  // pad for last PV k-step
    for (int k = L + lane; k < SEQ;  k += 32)
      __builtin_nontemporal_store(0.f, &arow[k]);             // exact zeros (ref underflows)
  }
  __syncthreads();

  // ---- context = P @ V, k-steps split across waves, reduced through LDS.
  // V is pre-transposed ([d][s]) so the B fragment is two contiguous 16 B runs per lane.
  v8f cacc[4] = {vzero8(), vzero8(), vzero8(), vzero8()};
  const int nsteps = Lpad >> 5;
  for (int s = wave; s < nsteps; s += 8) {
    int k0 = s * 32;
    v16bf ap;   // probs fp32 (LDS) -> bf16 A fragment
    {
      const float* p = scores + (lane & 15) * SEQ + k0 + ((lane & 16) ? 8 : 0);
#pragma unroll
      for (int v = 0; v < 8; ++v) {
        int k = (v < 4) ? 2 * v : 2 * v + 8;
        ap[2 * v]     = (bf16)p[k];
        ap[2 * v + 1] = (bf16)p[k + 1];
      }
    }
#pragma unroll
    for (int tn = 0; tn < 4; ++tn) {
      v16bf bv = b_frag(VbT, SEQ, tn * 16, k0, lane);   // B[k=seq][n=d]
      cacc[tn] = wmma_bf16f32(ap, bv, cacc[tn]);
    }
  }
  float* pw = partial + wave * (16 * 64);
#pragma unroll
  for (int tn = 0; tn < 4; ++tn)
#pragma unroll
    for (int r = 0; r < 8; ++r)
      pw[(mh + r) * 64 + tn * 16 + n] = cacc[tn][r];
  __syncthreads();

  for (int e = tid; e < 16 * 64; e += 256) {
    float ssum = 0.f;
#pragma unroll
    for (int w = 0; w < 8; ++w) ssum += partial[w * (16 * 64) + e];
    int m = e >> 6, d = e & 63;
    ctxOut[(size_t)(b * SEQ + q0 + m) * D_MODEL + h * DK + d] = ssum;
  }
}

// LayerNorm(proj + residual) over the last dim (512); one row per workgroup.
__global__ void lora_ln_kernel(const float* __restrict__ proj,
                               const float* __restrict__ resid,
                               float* __restrict__ out) {
  __shared__ float rs[8], rq[8];
  const int row = blockIdx.x, tid = threadIdx.x, lane = tid & 31, wave = tid >> 5;
  const float* p = proj + (size_t)row * D_MODEL;
  const float* r = resid + (size_t)row * D_MODEL;
  float x0 = p[tid] + r[tid];
  float x1 = p[tid + 256] + r[tid + 256];
  float s = x0 + x1, q = x0 * x0 + x1 * x1;
#pragma unroll
  for (int o = 16; o; o >>= 1) { s += __shfl_xor(s, o, 32); q += __shfl_xor(q, o, 32); }
  if (lane == 0) { rs[wave] = s; rq[wave] = q; }
  __syncthreads();
  float ts = 0.f, tq = 0.f;
#pragma unroll
  for (int w = 0; w < 8; ++w) { ts += rs[w]; tq += rq[w]; }
  float mean = ts * (1.f / D_MODEL);
  float var  = tq * (1.f / D_MODEL) - mean * mean;
  float inv  = rsqrtf(var + LN_EPS);
  out[(size_t)row * D_MODEL + tid]       = (x0 - mean) * inv;
  out[(size_t)row * D_MODEL + tid + 256] = (x1 - mean) * inv;
}

// ---------------------------------------------------------------- launch
extern "C" void kernel_launch(void* const* d_in, const int* in_sizes, int n_in,
                              void* d_out, int out_size, void* d_ws, size_t ws_size,
                              hipStream_t stream) {
  (void)in_sizes; (void)n_in; (void)out_size; (void)ws_size;
  const float* inQ = (const float*)d_in[0];
  const float* inK = (const float*)d_in[1];
  const float* inV = (const float*)d_in[2];
  // d_in[3] = attn_mask: causal triu, recomputed analytically in-kernel.
  const float* Wq = (const float*)d_in[4];
  const float* Wk = (const float*)d_in[5];
  const float* Wv = (const float*)d_in[6];
  const float* Aq = (const float*)d_in[7];
  const float* Bq = (const float*)d_in[8];
  const float* Ak = (const float*)d_in[9];
  const float* Bk = (const float*)d_in[10];
  const float* Av = (const float*)d_in[11];
  const float* Bv = (const float*)d_in[12];
  const float* Wo = (const float*)d_in[13];

  unsigned char* ws = (unsigned char*)d_ws;
  size_t off = 0;
  auto take = [&](size_t bytes) -> void* {
    void* p = ws + off;
    off += (bytes + 255) & ~(size_t)255;
    return p;
  };
  bf16* WqE  = (bf16*)take((size_t)D_MODEL * D_MODEL * sizeof(bf16));
  bf16* WkE  = (bf16*)take((size_t)D_MODEL * D_MODEL * sizeof(bf16));
  bf16* WvE  = (bf16*)take((size_t)D_MODEL * D_MODEL * sizeof(bf16));
  bf16* WoB  = (bf16*)take((size_t)D_MODEL * D_MODEL * sizeof(bf16));
  bf16* Qbf  = (bf16*)take((size_t)MTOT * D_MODEL * sizeof(bf16));
  bf16* Kbf  = (bf16*)take((size_t)MTOT * D_MODEL * sizeof(bf16));
  bf16* VbfT = (bf16*)take((size_t)MTOT * D_MODEL * sizeof(bf16));  // [b][d][s]
  float* ctx  = (float*)take((size_t)MTOT * D_MODEL * sizeof(float));
  float* proj = (float*)take((size_t)MTOT * D_MODEL * sizeof(float));

  float* outMain = (float*)d_out;                          // [B,S,512]
  float* attnOut = outMain + (size_t)MTOT * D_MODEL;       // [B,H,S,S]

  lora_prep_weights<<<(D_MODEL * D_MODEL + 255) / 256, 256, 0, stream>>>(
      Wq, Wk, Wv, Aq, Bq, Ak, Bk, Av, Bv, Wo, WqE, WkE, WvE, WoB);

  dim3 ggrid(MTOT / 64, D_MODEL / 128);
  // 1/sqrt(d_k) = 0.125 folded into Q at bf16 store.
  lora_gemm_xwT<<<ggrid, 256, 0, stream>>>(inQ, WqE, Qbf,  nullptr, 0, 0.125f);
  lora_gemm_xwT<<<ggrid, 256, 0, stream>>>(inK, WkE, Kbf,  nullptr, 0, 1.0f);
  lora_gemm_xwT<<<ggrid, 256, 0, stream>>>(inV, WvE, VbfT, nullptr, 2, 1.0f);

  const size_t smemBytes = (size_t)(16 * SEQ + 8 * 16 * 64) * sizeof(float); // 160 KB
  lora_attn_kernel<<<BATCH * NH * (SEQ / 16), 256, smemBytes, stream>>>(
      Qbf, Kbf, VbfT, attnOut, ctx);

  lora_gemm_xwT<<<ggrid, 256, 0, stream>>>(ctx, WoB, nullptr, proj, 1, 1.0f);
  lora_ln_kernel<<<MTOT, 256, 0, stream>>>(proj, inQ, outMain);
}
